// MultiHeadAttention_21990232555690
// MI455X (gfx1250) — matrix operand order repaired and checked by reading, not compiled, not run on hardware
//
#include <hip/hip_runtime.h>
#include <hip/hip_bf16.h>

// ---------------------------------------------------------------------------
// Problem constants (match reference)
// ---------------------------------------------------------------------------
constexpr int Bc  = 4;
constexpr int Sc  = 2048;
constexpr int Dc  = 1024;
constexpr int Hc  = 16;
constexpr int DKc = 64;
constexpr int Mrows = Bc * Sc;           // 8192 GEMM rows

// ---------------------------------------------------------------------------
// WMMA types (CDNA5 / gfx1250, wave32)
// ---------------------------------------------------------------------------
typedef __attribute__((ext_vector_type(16))) __bf16 v16bf;
typedef __attribute__((ext_vector_type(8)))  float  v8f;

union FragAB {
  v16bf v;
  uint4 q4[2];
  unsigned short u16[16];
};
union FragC {
  v8f  v;
  float f[8];
};

static __device__ __forceinline__ unsigned short f2bf(float x) {
  unsigned int u = __float_as_uint(x);
  u += 0x7fffu + ((u >> 16) & 1u);          // round-to-nearest-even
  return (unsigned short)(u >> 16);
}
static __device__ __forceinline__ float bf2f(unsigned short h) {
  return __uint_as_float(((unsigned int)h) << 16);
}

// ---------------------------------------------------------------------------
// Kernel 1: f32 -> bf16 conversion (grid-stride)
// ---------------------------------------------------------------------------
__global__ void cvt_f32_to_bf16(const float* __restrict__ in,
                                unsigned short* __restrict__ out, int n) {
  int i = blockIdx.x * blockDim.x + threadIdx.x;
  int stride = gridDim.x * blockDim.x;
  for (; i < n; i += stride) out[i] = f2bf(in[i]);
}

// ---------------------------------------------------------------------------
// Kernel 2: WMMA GEMM  out[M,N] = A[M,K](bf16) * Bm[N,K]^T(bf16) + bias[N]
// Block = 128 threads (4 waves), block tile 64x64, wave tile 32x32.
// __launch_bounds__(128,1): let the compiler keep all fragments in VGPRs.
// ---------------------------------------------------------------------------
template <bool F32OUT>
__global__ void __launch_bounds__(128, 1)
gemm_bias_wmma(const unsigned short* __restrict__ A,
               const unsigned short* __restrict__ Bm,
               const float* __restrict__ bias,
               void* __restrict__ out,
               int Ndim, int Kdim) {
  const int lane   = threadIdx.x & 31;
  const int wave   = threadIdx.x >> 5;
  const int lane16 = lane & 15;
  const int half   = lane >> 4;

  const int m0 = blockIdx.y * 64 + (wave >> 1) * 32;
  const int n0 = blockIdx.x * 64 + (wave & 1) * 32;

  FragC acc[2][2];
#pragma unroll
  for (int i = 0; i < 2; ++i)
#pragma unroll
    for (int j = 0; j < 2; ++j)
#pragma unroll
      for (int r = 0; r < 8; ++r) acc[i][j].f[r] = 0.0f;

  const int ca = half ? 8 : 0;    // A-frag K offset for this half-wave
  const int cb = half ? 8 : 0;    // B-frag K offset for this half-wave

  for (int k0 = 0; k0 < Kdim; k0 += 32) {
    FragAB a[2], b[2];
#pragma unroll
    for (int i = 0; i < 2; ++i) {
      const unsigned short* p =
          A + (size_t)(m0 + i * 16 + lane16) * Kdim + k0 + ca;
      a[i].q4[0] = *(const uint4*)p;          // K = c .. c+7
      a[i].q4[1] = *(const uint4*)(p + 16);   // K = c+16 .. c+23
    }
#pragma unroll
    for (int j = 0; j < 2; ++j) {
      const unsigned short* p =
          Bm + (size_t)(n0 + j * 16 + lane16) * Kdim + k0 + cb;
      b[j].q4[0] = *(const uint4*)p;          // K = cc .. cc+7
      b[j].q4[1] = *(const uint4*)(p + 16);   // K = cc+16 .. cc+23
    }
#pragma unroll
    for (int i = 0; i < 2; ++i)
#pragma unroll
      for (int j = 0; j < 2; ++j)
        acc[i][j].v = __builtin_amdgcn_wmma_f32_16x16x32_bf16(
            false, a[i].v, false, b[j].v, (short)0, acc[i][j].v, false, false);
  }

#pragma unroll
  for (int j = 0; j < 2; ++j) {
    const int n = n0 + j * 16 + lane16;
    const float bv = bias[n];
#pragma unroll
    for (int i = 0; i < 2; ++i) {
#pragma unroll
      for (int r = 0; r < 8; ++r) {
        const int m = m0 + i * 16 + half * 8 + r;
        const float val = acc[i][j].f[r] + bv;
        if (F32OUT)
          ((float*)out)[(size_t)m * Ndim + n] = val;
        else
          ((unsigned short*)out)[(size_t)m * Ndim + n] = f2bf(val);
      }
    }
  }
}

// ---------------------------------------------------------------------------
// Kernel 3: RoPE + repack.
//   Q0/K0/V0: bf16 [B,S,D]  (D = H*DK, head-interleaved)
//   Qb,Kb   : bf16 [B,H,S,DK]   (Q pre-scaled by 1/sqrt(DK))
//   Vt      : bf16 [B,H,DK,S]   (transposed for PV B-fragments)
// One thread per (b,s,pair).
// ---------------------------------------------------------------------------
__global__ void rope_pack(const unsigned short* __restrict__ Q0,
                          const unsigned short* __restrict__ K0,
                          const unsigned short* __restrict__ V0,
                          const float* __restrict__ fcos,
                          const float* __restrict__ fsin,
                          unsigned short* __restrict__ Qb,
                          unsigned short* __restrict__ Kb,
                          unsigned short* __restrict__ Vt) {
  const int idx = blockIdx.x * blockDim.x + threadIdx.x;
  constexpr int pairsPerRow = Dc / 2;                 // 512
  constexpr int total = Bc * Sc * pairsPerRow;
  if (idx >= total) return;

  const int ph = idx % pairsPerRow;
  const int s  = (idx / pairsPerRow) % Sc;
  const int b  = idx / (pairsPerRow * Sc);
  const int h  = ph / (DKc / 2);
  const int i  = ph % (DKc / 2);
  const int dk = 2 * i;

  const size_t src = ((size_t)(b * Sc + s)) * Dc + h * DKc + dk;
  const float c  = fcos[s * (DKc / 2) + i];
  const float sn = fsin[s * (DKc / 2) + i];

  const size_t dstQK = (((size_t)(b * Hc + h)) * Sc + s) * DKc + dk;

  // Q : rope + 1/sqrt(DK) scale folded in
  {
    const unsigned int p = *(const unsigned int*)(Q0 + src);
    const float x1 = bf2f((unsigned short)(p & 0xffffu));
    const float x2 = bf2f((unsigned short)(p >> 16));
    const float r1 = (x1 * c - x2 * sn) * 0.125f;
    const float r2 = (x1 * sn + x2 * c) * 0.125f;
    *(unsigned int*)(Qb + dstQK) =
        (unsigned int)f2bf(r1) | ((unsigned int)f2bf(r2) << 16);
  }
  // K : rope
  {
    const unsigned int p = *(const unsigned int*)(K0 + src);
    const float x1 = bf2f((unsigned short)(p & 0xffffu));
    const float x2 = bf2f((unsigned short)(p >> 16));
    const float r1 = x1 * c - x2 * sn;
    const float r2 = x1 * sn + x2 * c;
    *(unsigned int*)(Kb + dstQK) =
        (unsigned int)f2bf(r1) | ((unsigned int)f2bf(r2) << 16);
  }
  // V : transpose copy (bit-exact)
  {
    const unsigned int p = *(const unsigned int*)(V0 + src);
    const size_t base = ((size_t)(b * Hc + h)) * DKc * Sc;
    Vt[base + (size_t)dk * Sc + s]       = (unsigned short)(p & 0xffffu);
    Vt[base + (size_t)(dk + 1) * Sc + s] = (unsigned short)(p >> 16);
  }
}

// ---------------------------------------------------------------------------
// Kernel 4: causal flash attention.
//   Qb,Kb: bf16 [B,H,S,DK], Vt: bf16 [B,H,DK,S] -> Ob: bf16 [B,S,D]
// Block = 128 threads (4 waves); each wave owns a 16-row q tile; block
// covers 64 q rows so the K loop bound is uniform across the block.
// __launch_bounds__(128,1): ~160 live VGPRs must stay resident (no spill).
// ---------------------------------------------------------------------------
__global__ void __launch_bounds__(128, 1)
attn_fwd(const unsigned short* __restrict__ Qb,
         const unsigned short* __restrict__ Kb,
         const unsigned short* __restrict__ Vt,
         unsigned short* __restrict__ Ob) {
  __shared__ float plds[4][16][33];   // per-wave P transpose staging

  const int lane   = threadIdx.x & 31;
  const int wave   = threadIdx.x >> 5;
  const int lane16 = lane & 15;
  const int half   = lane >> 4;

  const int bh   = blockIdx.y;           // b*H + h
  const int b    = bh / Hc;
  const int h    = bh % Hc;
  const int qblk = blockIdx.x * 64;
  const int q0   = qblk + wave * 16;

  const unsigned short* Qh = Qb + (size_t)bh * Sc * DKc;
  const unsigned short* Kh = Kb + (size_t)bh * Sc * DKc;
  const unsigned short* Vh = Vt + (size_t)bh * DKc * Sc;

  // Q A-fragments for dk 0..31 and 32..63 (persistent across K loop)
  FragAB qf[2];
  {
    const int ca = half ? 8 : 0;
    const unsigned short* row = Qh + (size_t)(q0 + lane16) * DKc;
#pragma unroll
    for (int f = 0; f < 2; ++f) {
      const unsigned short* p = row + f * 32 + ca;
      qf[f].q4[0] = *(const uint4*)p;
      qf[f].q4[1] = *(const uint4*)(p + 16);
    }
  }

  FragC o[4];
#pragma unroll
  for (int d = 0; d < 4; ++d)
#pragma unroll
    for (int r = 0; r < 8; ++r) o[d].f[r] = 0.0f;

  float mrow[8], lrow[8];
#pragma unroll
  for (int r = 0; r < 8; ++r) { mrow[r] = -3.0e38f; lrow[r] = 0.0f; }

  const int kend = qblk + 64;            // causal: uniform across block

  for (int kb = 0; kb < kend; kb += 32) {
    // ---- scores: two 16x16 tiles covering kpos kb..kb+31 --------------
    FragC sc[2];
    {
      const int cb = half ? 8 : 0;
#pragma unroll
      for (int t = 0; t < 2; ++t) {
#pragma unroll
        for (int r = 0; r < 8; ++r) sc[t].f[r] = 0.0f;
        const unsigned short* pr =
            Kh + (size_t)(kb + t * 16 + lane16) * DKc;
#pragma unroll
        for (int f = 0; f < 2; ++f) {
          FragAB bk;
          const unsigned short* p = pr + f * 32 + cb;
          bk.q4[0] = *(const uint4*)p;
          bk.q4[1] = *(const uint4*)(p + 16);
          sc[t].v = __builtin_amdgcn_wmma_f32_16x16x32_bf16(
              false, qf[f].v, false, bk.v, (short)0, sc[t].v, false, false);
        }
      }
    }

    // ---- causal mask + online softmax (rows live across 16-lane halves)
#pragma unroll
    for (int r = 0; r < 8; ++r) {
      const int q = q0 + half * 8 + r;
      float mx = -3.0e38f;
#pragma unroll
      for (int t = 0; t < 2; ++t) {
        const int k = kb + t * 16 + lane16;
        if (k > q) sc[t].f[r] = -3.0e38f;
        mx = fmaxf(mx, sc[t].f[r]);
      }
#pragma unroll
      for (int off = 1; off < 16; off <<= 1)
        mx = fmaxf(mx, __shfl_xor(mx, off));
      const float mnew = fmaxf(mrow[r], mx);
      const float alpha = __expf(mrow[r] - mnew);
      float sum = 0.0f;
#pragma unroll
      for (int t = 0; t < 2; ++t) {
        const float p = __expf(sc[t].f[r] - mnew);
        sc[t].f[r] = p;
        sum += p;
      }
#pragma unroll
      for (int off = 1; off < 16; off <<= 1) sum += __shfl_xor(sum, off);
      lrow[r] = lrow[r] * alpha + sum;
      mrow[r] = mnew;
#pragma unroll
      for (int d = 0; d < 4; ++d) o[d].f[r] *= alpha;
    }

    // ---- transpose P (C-layout -> A-layout) through LDS ---------------
#pragma unroll
    for (int t = 0; t < 2; ++t)
#pragma unroll
      for (int r = 0; r < 8; ++r)
        plds[wave][half * 8 + r][t * 16 + lane16] = sc[t].f[r];
    __syncthreads();   // uniform trip count across the 4 waves

    FragAB pf;
    {
      const int cp = half ? 8 : 0;
#pragma unroll
      for (int e = 0; e < 8; ++e) {
        pf.u16[e]     = f2bf(plds[wave][lane16][cp + e]);
        pf.u16[8 + e] = f2bf(plds[wave][lane16][cp + 16 + e]);
      }
    }

    // ---- PV: o[d] += P(16x32) * V^T-fragment --------------------------
    {
      const int cb = half ? 8 : 0;
#pragma unroll
      for (int d = 0; d < 4; ++d) {
        FragAB bv;
        const unsigned short* p =
            Vh + (size_t)(d * 16 + lane16) * Sc + kb + cb;
        bv.q4[0] = *(const uint4*)p;
        bv.q4[1] = *(const uint4*)(p + 16);
        o[d].v = __builtin_amdgcn_wmma_f32_16x16x32_bf16(
            false, pf.v, false, bv.v, (short)0, o[d].v, false, false);
      }
    }
  }

  // ---- normalize + store O as bf16 [B,S,D] ----------------------------
#pragma unroll
  for (int r = 0; r < 8; ++r) {
    const float inv = 1.0f / lrow[r];
    const int q = q0 + half * 8 + r;
#pragma unroll
    for (int d = 0; d < 4; ++d) {
      const int col = h * DKc + d * 16 + lane16;
      Ob[((size_t)(b * Sc + q)) * Dc + col] = f2bf(o[d].f[r] * inv);
    }
  }
}

// ---------------------------------------------------------------------------
// Host-side orchestration
// ---------------------------------------------------------------------------
extern "C" void kernel_launch(void* const* d_in, const int* in_sizes, int n_in,
                              void* d_out, int out_size, void* d_ws,
                              size_t ws_size, hipStream_t stream) {
  const float* x   = (const float*)d_in[0];
  const float* W_q = (const float*)d_in[1];
  const float* b_q = (const float*)d_in[2];
  const float* W_k = (const float*)d_in[3];
  const float* b_k = (const float*)d_in[4];
  const float* W_v = (const float*)d_in[5];
  const float* b_v = (const float*)d_in[6];
  const float* W_o = (const float*)d_in[7];
  const float* b_o = (const float*)d_in[8];
  const float* fc  = (const float*)d_in[9];
  const float* fs  = (const float*)d_in[10];
  // d_in[11] = mask: causal, handled analytically.

  // Workspace layout (bytes). Total = 120 MB.
  char* ws = (char*)d_ws;
  const size_t xbytes = (size_t)Mrows * Dc * 2;   // 16 MB
  const size_t wbytes = (size_t)Dc * Dc * 2;      //  2 MB
  unsigned short* xb  = (unsigned short*)(ws);
  unsigned short* wqb = (unsigned short*)(ws + xbytes);
  unsigned short* wkb = (unsigned short*)(ws + xbytes + wbytes);
  unsigned short* wvb = (unsigned short*)(ws + xbytes + 2 * wbytes);
  unsigned short* wob = (unsigned short*)(ws + xbytes + 3 * wbytes);
  unsigned short* Q0  = (unsigned short*)(ws + xbytes + 4 * wbytes);
  unsigned short* K0  = Q0 + (size_t)Mrows * Dc;
  unsigned short* V0  = K0 + (size_t)Mrows * Dc;
  unsigned short* Qb  = V0 + (size_t)Mrows * Dc;
  unsigned short* Kb  = Qb + (size_t)Mrows * Dc;
  unsigned short* Vt  = Kb + (size_t)Mrows * Dc;
  unsigned short* Ob  = Q0;   // Q0 is dead after rope_pack; reuse its region.

  // 1) f32 -> bf16 conversions
  {
    const int nx = Mrows * Dc;
    cvt_f32_to_bf16<<<(nx + 255) / 256, 256, 0, stream>>>(x, xb, nx);
    const int nw = Dc * Dc;
    cvt_f32_to_bf16<<<(nw + 255) / 256, 256, 0, stream>>>(W_q, wqb, nw);
    cvt_f32_to_bf16<<<(nw + 255) / 256, 256, 0, stream>>>(W_k, wkb, nw);
    cvt_f32_to_bf16<<<(nw + 255) / 256, 256, 0, stream>>>(W_v, wvb, nw);
    cvt_f32_to_bf16<<<(nw + 255) / 256, 256, 0, stream>>>(W_o, wob, nw);
  }

  // 2) QKV projections (bf16 out + bias)
  {
    dim3 grid(Dc / 64, Mrows / 64);     // (16, 128)
    dim3 block(128);
    gemm_bias_wmma<false><<<grid, block, 0, stream>>>(xb, wqb, b_q, Q0, Dc, Dc);
    gemm_bias_wmma<false><<<grid, block, 0, stream>>>(xb, wkb, b_k, K0, Dc, Dc);
    gemm_bias_wmma<false><<<grid, block, 0, stream>>>(xb, wvb, b_v, V0, Dc, Dc);
  }

  // 3) RoPE + repack (Q scaled by 1/sqrt(DK); V transposed)
  {
    const int total = Bc * Sc * (Dc / 2);
    rope_pack<<<(total + 255) / 256, 256, 0, stream>>>(Q0, K0, V0, fc, fs,
                                                       Qb, Kb, Vt);
  }

  // 4) causal flash attention
  {
    dim3 grid(Sc / 64, Bc * Hc);        // (32, 64)
    dim3 block(128);
    attn_fwd<<<grid, block, 0, stream>>>(Qb, Kb, Vt, Ob);
  }

  // 5) output projection (f32 out + bias) -> d_out
  {
    dim3 grid(Dc / 64, Mrows / 64);
    dim3 block(128);
    gemm_bias_wmma<true><<<grid, block, 0, stream>>>(Ob, wob, b_o,
                                                     (float*)d_out, Dc, Dc);
  }
}
